// SelectiveLinear_17033840296197
// MI455X (gfx1250) — compile-verified, matrix operand order checked by reading, not run on hardware
//
#include <hip/hip_runtime.h>

#define N_OPT 16
#define OUT_F 1024
#define IN_F  1024
#define BATCH 8
#define AROWS 512

typedef __attribute__((ext_vector_type(8)))  __bf16 v8bf;   // 16 bytes
typedef __attribute__((ext_vector_type(16))) __bf16 v16bf;  // 32 bytes (8 VGPRs)
typedef __attribute__((ext_vector_type(8)))  float  v8f;    // 8 VGPRs

// ---------- bf16 helpers ----------
__device__ __forceinline__ float bf16_to_f32(__bf16 h) {
    unsigned short s = __builtin_bit_cast(unsigned short, h);
    union { unsigned u; float f; } c; c.u = ((unsigned)s) << 16;
    return c.f;
}
__device__ __forceinline__ __bf16 f32_to_bf16_rne(float f) {
    union { float f; unsigned u; } c; c.f = f;
    unsigned r = c.u + 0x7FFFu + ((c.u >> 16) & 1u);
    unsigned short s = (unsigned short)(r >> 16);
    return __builtin_bit_cast(__bf16, s);
}
// split f32 ~= hi + lo (both bf16); hi*hi + hi*lo + lo*hi recovers ~16 mantissa bits
__device__ __forceinline__ void split_bf16(float v, __bf16& hi, __bf16& lo) {
    hi = f32_to_bf16_rne(v);
    lo = f32_to_bf16_rne(v - bf16_to_f32(hi));
}

// ---------- kernel 1: mix weights across options, emit split-bf16 ----------
__global__ __launch_bounds__(256) void mix_split_weights_kernel(
    const float* __restrict__ weight,   // [N_OPT][OUT_F][IN_F]
    const float* __restrict__ sp,       // [BATCH][N_OPT]
    __bf16* __restrict__ whi, __bf16* __restrict__ wlo) {
    __shared__ float s_sp[BATCH * N_OPT];
    if (threadIdx.x < BATCH * N_OPT) s_sp[threadIdx.x] = sp[threadIdx.x];
    __syncthreads();

    const size_t idx = (size_t)blockIdx.x * blockDim.x + threadIdx.x; // over OUT_F*IN_F
    float acc[BATCH];
#pragma unroll
    for (int b = 0; b < BATCH; ++b) acc[b] = 0.0f;

#pragma unroll
    for (int n = 0; n < N_OPT; ++n) {
        const float w = weight[(size_t)n * (OUT_F * IN_F) + idx];
#pragma unroll
        for (int b = 0; b < BATCH; ++b)
            acc[b] = __builtin_fmaf(s_sp[b * N_OPT + n], w, acc[b]);
    }
#pragma unroll
    for (int b = 0; b < BATCH; ++b) {
        __bf16 hi, lo;
        split_bf16(acc[b], hi, lo);
        whi[(size_t)b * (OUT_F * IN_F) + idx] = hi;
        wlo[(size_t)b * (OUT_F * IN_F) + idx] = lo;
    }
}

// ---------- kernel 2: split x into bf16 hi/lo ----------
__global__ __launch_bounds__(256) void split_x_kernel(
    const float* __restrict__ x, __bf16* __restrict__ xhi, __bf16* __restrict__ xlo) {
    const size_t idx = (size_t)blockIdx.x * blockDim.x + threadIdx.x;
    __bf16 hi, lo;
    split_bf16(x[idx], hi, lo);
    xhi[idx] = hi;
    xlo[idx] = lo;
}

// ---------- kernel 3: mix bias ----------
__global__ __launch_bounds__(256) void mix_bias_kernel(
    const float* __restrict__ bias,     // [N_OPT][OUT_F]
    const float* __restrict__ sp,       // [BATCH][N_OPT]
    float* __restrict__ bmix) {         // [BATCH][OUT_F]
    const int idx = blockIdx.x * blockDim.x + threadIdx.x;   // over BATCH*OUT_F
    const int b = idx >> 10;
    const int i = idx & (OUT_F - 1);
    float acc = 0.0f;
#pragma unroll
    for (int n = 0; n < N_OPT; ++n)
        acc = __builtin_fmaf(sp[b * N_OPT + n], bias[n * OUT_F + i], acc);
    bmix[idx] = acc;
}

// ---------- CDNA5 async copy: global -> LDS (tracked by ASYNCcnt) ----------
__device__ __forceinline__ void async_b128(unsigned lds_addr, const void* gptr) {
    unsigned long long ga = (unsigned long long)gptr;
    asm volatile("global_load_async_to_lds_b128 %0, %1, off"
                 :: "v"(lds_addr), "v"(ga) : "memory");
}
__device__ __forceinline__ void wait_async0() {
    asm volatile("s_wait_asynccnt 0x0" ::: "memory");
}

// ---------- WMMA helpers ----------
__device__ __forceinline__ v16bf load_frag_lds(const char* p) {
    v8bf lo = *(const v8bf*)(p);         // K = kh+0..kh+7
    v8bf hi = *(const v8bf*)(p + 32);    // K = kh+16..kh+23
    return __builtin_shufflevector(lo, hi, 0, 1, 2, 3, 4, 5, 6, 7,
                                   8, 9, 10, 11, 12, 13, 14, 15);
}
__device__ __forceinline__ v8f wmma_bf16(v16bf a, v16bf b, v8f c) {
    return __builtin_amdgcn_wmma_f32_16x16x32_bf16(false, a, false, b,
                                                   (short)0, c, false, false);
}

// ---------- WMMA GEMM, LDS double-buffered via async copy ----------
// block = 256 threads (8 waves, 2Mx4N), block tile 64(M) x 128(N), K slab = 32.
// LDS buffer layout (row stride 80B = 64B data + 16B anti-conflict pad):
//   A_hi [64 rows]  @ 0       (5120 B)
//   A_lo [64 rows]  @ 5120    (5120 B)
//   B_hi [128 rows] @ 10240   (10240 B)
//   B_lo [128 rows] @ 20480   (10240 B)
// buffer size 30720 B, double buffered = 61440 B.
#define ROWB   80
#define A_LO   5120
#define B_HI   10240
#define B_LO   20480
#define BUFSZ  30720

__global__ __launch_bounds__(256) void selgemm_kernel(
    const __bf16* __restrict__ xh, const __bf16* __restrict__ xl,   // [B][A][IN]
    const __bf16* __restrict__ wh, const __bf16* __restrict__ wl,   // [B][OUT][IN]
    const float* __restrict__ bmix,                                 // [B][OUT]
    float* __restrict__ out) {                                      // [B][A][OUT]
    __shared__ __align__(128) char smem[2 * BUFSZ];

    const int t    = threadIdx.x;
    const int lane = t & 31;
    const int wave = t >> 5;
    const int lrow = lane & 15;
    const int khB  = (lane >> 4) << 4;           // K-half byte offset: 0 or 16
    const int b    = blockIdx.z;
    const int m_blk = blockIdx.x * 64;
    const int n_blk = blockIdx.y * 128;

    // ---- producer (async copy) mapping: 256 threads x 16B cover one 64-row slab
    const int prow   = t >> 2;                   // 0..63
    const int pchunk = t & 3;                    // 4 x 16B chunks per 64B row
    const char* gA_h = (const char*)(xh + (size_t)(b * AROWS + m_blk + prow) * IN_F + pchunk * 8);
    const char* gA_l = (const char*)(xl + (size_t)(b * AROWS + m_blk + prow) * IN_F + pchunk * 8);
    const char* gB_h0 = (const char*)(wh + (size_t)(b * OUT_F + n_blk + prow) * IN_F + pchunk * 8);
    const char* gB_l0 = (const char*)(wl + (size_t)(b * OUT_F + n_blk + prow) * IN_F + pchunk * 8);
    const char* gB_h1 = gB_h0 + (size_t)64 * IN_F * 2;
    const char* gB_l1 = gB_l0 + (size_t)64 * IN_F * 2;

    const unsigned smem_base = (unsigned)(size_t)(void*)smem;      // low 32 bits = LDS byte addr
    const unsigned pA = (unsigned)(prow * ROWB + pchunk * 16);
    const unsigned pB = pA;                                        // same shape per 64-row group

    // ---- consumer (fragment) LDS offsets
    const int m_w = (wave & 1) * 32;             // wave M offset in block tile
    const int n_w = (wave >> 1) * 32;            // wave N offset in block tile
    const unsigned a0 = (unsigned)((m_w + lrow) * ROWB + khB);
    const unsigned a1 = (unsigned)((m_w + 16 + lrow) * ROWB + khB);
    const unsigned b0 = (unsigned)(B_HI + (n_w + lrow) * ROWB + khB);
    const unsigned b1 = (unsigned)(B_HI + (n_w + 16 + lrow) * ROWB + khB);

    v8f acc00 = {}, acc01 = {}, acc10 = {}, acc11 = {};

    // prologue: stage K-slab 0 into buffer 0
    {
        const unsigned bb = smem_base;
        async_b128(bb + pA,                gA_h);
        async_b128(bb + A_LO + pA,         gA_l);
        async_b128(bb + B_HI + pB,         gB_h0);
        async_b128(bb + B_HI + 5120 + pB,  gB_h1);
        async_b128(bb + B_LO + pB,         gB_l0);
        async_b128(bb + B_LO + 5120 + pB,  gB_l1);
    }

#pragma unroll 1
    for (int ks = 0; ks < IN_F / 32; ++ks) {
        wait_async0();          // my slab pieces have landed in LDS
        __syncthreads();        // everyone's landed; previous buffer fully consumed

        if (ks + 1 < IN_F / 32) {
            const int kbyte = (ks + 1) * 64;                 // 32 K-elems = 64 B
            const unsigned bb = smem_base + ((ks + 1) & 1) * BUFSZ;
            async_b128(bb + pA,               gA_h + kbyte);
            async_b128(bb + A_LO + pA,        gA_l + kbyte);
            async_b128(bb + B_HI + pB,        gB_h0 + kbyte);
            async_b128(bb + B_HI + 5120 + pB, gB_h1 + kbyte);
            async_b128(bb + B_LO + pB,        gB_l0 + kbyte);
            async_b128(bb + B_LO + 5120 + pB, gB_l1 + kbyte);
        }

        const char* buf = smem + (ks & 1) * BUFSZ;
        const v16bf a0h = load_frag_lds(buf + a0);
        const v16bf a0l = load_frag_lds(buf + A_LO + a0);
        const v16bf a1h = load_frag_lds(buf + a1);
        const v16bf a1l = load_frag_lds(buf + A_LO + a1);
        const v16bf b0h = load_frag_lds(buf + b0);
        const v16bf b0l = load_frag_lds(buf + (B_LO - B_HI) + b0);
        const v16bf b1h = load_frag_lds(buf + b1);
        const v16bf b1l = load_frag_lds(buf + (B_LO - B_HI) + b1);

        acc00 = wmma_bf16(a0h, b0h, acc00);
        acc00 = wmma_bf16(a0h, b0l, acc00);
        acc00 = wmma_bf16(a0l, b0h, acc00);

        acc01 = wmma_bf16(a0h, b1h, acc01);
        acc01 = wmma_bf16(a0h, b1l, acc01);
        acc01 = wmma_bf16(a0l, b1h, acc01);

        acc10 = wmma_bf16(a1h, b0h, acc10);
        acc10 = wmma_bf16(a1h, b0l, acc10);
        acc10 = wmma_bf16(a1l, b0h, acc10);

        acc11 = wmma_bf16(a1h, b1h, acc11);
        acc11 = wmma_bf16(a1h, b1l, acc11);
        acc11 = wmma_bf16(a1l, b1h, acc11);
    }

    // Epilogue: C/D layout -> VGPR r, lanes 0-15: (M=r, N=lane); lanes 16-31: (M=r+8, N=lane-16)
    const int n_base = n_blk + n_w;
    const int m_base = m_blk + m_w;
    const float bias0 = bmix[b * OUT_F + n_base + lrow];
    const float bias1 = bmix[b * OUT_F + n_base + 16 + lrow];
    const int   rbase = m_base + ((lane >> 4) << 3);
    float* __restrict__ outb = out + (size_t)b * AROWS * OUT_F;

#pragma unroll
    for (int r = 0; r < 8; ++r) {
        const int row0 = rbase + r;
        const int row1 = rbase + 16 + r;
        outb[(size_t)row0 * OUT_F + n_base + lrow]      = acc00[r] + bias0;
        outb[(size_t)row0 * OUT_F + n_base + 16 + lrow] = acc01[r] + bias1;
        outb[(size_t)row1 * OUT_F + n_base + lrow]      = acc10[r] + bias0;
        outb[(size_t)row1 * OUT_F + n_base + 16 + lrow] = acc11[r] + bias1;
    }
}

extern "C" void kernel_launch(void* const* d_in, const int* in_sizes, int n_in,
                              void* d_out, int out_size, void* d_ws, size_t ws_size,
                              hipStream_t stream) {
    const float* x      = (const float*)d_in[0];  // [8][512][1024]
    const float* sp     = (const float*)d_in[1];  // [8][16]
    const float* weight = (const float*)d_in[2];  // [16][1024][1024]
    const float* bias   = (const float*)d_in[3];  // [16][1024]
    float* out          = (float*)d_out;          // [8][512][1024]

    char* ws = (char*)d_ws;
    const size_t MB = 1024 * 1024;
    __bf16* whi = (__bf16*)(ws);                  // 16 MB
    __bf16* wlo = (__bf16*)(ws + 16 * MB);        // 16 MB
    __bf16* xhi = (__bf16*)(ws + 32 * MB);        //  8 MB
    __bf16* xlo = (__bf16*)(ws + 40 * MB);        //  8 MB
    float*  bmx = (float*)(ws + 48 * MB);         // 32 KB

    mix_split_weights_kernel<<<(OUT_F * IN_F) / 256, 256, 0, stream>>>(weight, sp, whi, wlo);
    split_x_kernel<<<(BATCH * AROWS * IN_F) / 256, 256, 0, stream>>>(x, xhi, xlo);
    mix_bias_kernel<<<(BATCH * OUT_F) / 256, 256, 0, stream>>>(bias, sp, bmx);

    dim3 grid(AROWS / 64, OUT_F / 128, BATCH);
    selgemm_kernel<<<grid, 256, 0, stream>>>(xhi, xlo, whi, wlo, bmx, out);
}
